// TritonRNN_33122787787506
// MI455X (gfx1250) — matrix operand order repaired and checked by reading, not compiled, not run on hardware
//
#include <hip/hip_runtime.h>
#include <hip/hip_bf16.h>

// ---------------- types ----------------
typedef __bf16 bf16;
typedef __attribute__((ext_vector_type(16))) bf16  bf16x16;
typedef __attribute__((ext_vector_type(8)))  float f32x8;
typedef __attribute__((ext_vector_type(4)))  float f32x4;
typedef __attribute__((ext_vector_type(4)))  unsigned int u32x4;

union Frag32B { u32x4 q[2]; bf16x16 v; };

// D = A(16x32 bf16) * B(32x16 bf16) + C(16x16 f32)
static __device__ __forceinline__ f32x8 wmma_bf16(bf16x16 a, bf16x16 b, f32x8 c) {
  return __builtin_amdgcn_wmma_f32_16x16x32_bf16(
      /*neg_a=*/false, a, /*neg_b=*/false, b,
      /*c_mod=*/(short)0, c, /*reuse_a=*/false, /*reuse_b=*/false);
}

// Fragment loads: uniform base + 32-bit byte offset -> SADDR-form global loads.
// A-frag: gap=32B (k 0..7 then k 16..23 per lane-half per ISA A layout)
// B-frag: gap=32B (same k order as the A-frag)
static __device__ __forceinline__ bf16x16 ldfrag_g(const void* base, unsigned byteOff,
                                                   unsigned gapBytes) {
  const char* p = (const char*)base + (size_t)byteOff;
  Frag32B f;
  f.q[0] = *(const u32x4*)(p);
  f.q[1] = *(const u32x4*)(p + gapBytes);
  return f.v;
}

// LDS fragment load (32-bit LDS addressing, plain pointer form)
static __device__ __forceinline__ bf16x16 ldfrag_lds(const bf16* p, int gapElems) {
  Frag32B f;
  f.q[0] = *(const u32x4*)(p);
  f.q[1] = *(const u32x4*)(p + gapElems);
  return f.v;
}

// Split-bf16 conversion of 8 floats at base+byteOff: hi = bf16(x), lo = bf16(x - hi)
static __device__ __forceinline__ void split8_g(const void* base, unsigned byteOff,
                                                int vbase, bf16x16& hi, bf16x16& lo) {
  const char* p = (const char*)base + (size_t)byteOff;
  #pragma unroll
  for (int c = 0; c < 2; ++c) {
    f32x4 v = *(const f32x4*)(p + 16 * c);
    #pragma unroll
    for (int j = 0; j < 4; ++j) {
      float f = v[j];
      bf16  h = (bf16)f;
      hi[vbase + 4 * c + j] = h;
      lo[vbase + 4 * c + j] = (bf16)(f - (float)h);
    }
  }
}

static __device__ __forceinline__ f32x8 zero8() {
  f32x8 z;
  #pragma unroll
  for (int i = 0; i < 8; ++i) z[i] = 0.0f;
  return z;
}

// ---------------- phase 0: one-shot W_ih -> split-bf16 planes ----------------
__global__ void __launch_bounds__(256)
rnn_cvt_wih_kernel(const float* __restrict__ w, unsigned short* __restrict__ whi,
                   unsigned short* __restrict__ wlo) {
  const unsigned i4 = ((unsigned)blockIdx.x * 256u + threadIdx.x) * 4u;
  f32x4 v = *(const f32x4*)(w + i4);
  bf16* dh = (bf16*)whi + i4;
  bf16* dl = (bf16*)wlo + i4;
  #pragma unroll
  for (int j = 0; j < 4; ++j) {
    bf16 h = (bf16)v[j];
    dh[j] = h;
    dl[j] = (bf16)(v[j] - (float)h);
  }
}

// ---------------- phase 1: x_proj = x @ W_ih^T + b_ih + b_hh ----------------
// grid (8 n-strips, 128 m-groups), block 256 (8 waves).
// Each wave: one 16(M) x 128(N) strip, K=1024 in 32-chunks, split-bf16
// (3 WMMAs/tile, 24 WMMAs per K-chunk). B fragments loaded in groups of 4
// N-tiles into distinct registers -> >=8 loads in flight, staggered waits.
#define PROJ_NT 8
__global__ void __launch_bounds__(256)
rnn_proj_kernel(const float* __restrict__ x,
                const unsigned short* __restrict__ wih_hi,
                const unsigned short* __restrict__ wih_lo,
                const float* __restrict__ bias_ih, const float* __restrict__ bias_hh,
                float* __restrict__ out) {
  const unsigned lane = threadIdx.x & 31;
  const unsigned wave = threadIdx.x >> 5;
  const unsigned lh   = lane >> 4;   // lane-half
  const unsigned lm   = lane & 15;   // row/col within tile
  const unsigned mtile = blockIdx.y * 8 + wave;
  const unsigned mbase = mtile * 16;
  const unsigned nbase = blockIdx.x * (PROJ_NT * 16);

  f32x8 acc[PROJ_NT];
  #pragma unroll
  for (int t = 0; t < PROJ_NT; ++t) acc[t] = zero8();

  // byte offsets (all lane-varying addressing stays 32-bit -> SADDR form)
  const unsigned xRowB = (mbase + lm) << 12;                // *1024 floats *4B
  #pragma unroll 1
  for (int kc = 0; kc < 32; ++kc) {
    const unsigned kb  = (unsigned)kc * 32u;
    const unsigned aB0 = xRowB + ((kb +      8u * lh) << 2);
    const unsigned aB1 = xRowB + ((kb + 16 + 8u * lh) << 2);
    bf16x16 ahi, alo;
    split8_g(x, aB0, 0, ahi, alo);
    split8_g(x, aB1, 8, ahi, alo);
    const unsigned kByte = (kb + 8u * lh) << 1;
    #pragma unroll
    for (int g = 0; g < 2; ++g) {
      bf16x16 bh[4], bl[4];
      #pragma unroll
      for (int j = 0; j < 4; ++j) {
        const unsigned nt = (unsigned)(g * 4 + j);
        const unsigned wB = ((nbase + nt * 16u + lm) << 11) + kByte;  // *1024*2B
        bh[j] = ldfrag_g(wih_hi, wB, 32);
        bl[j] = ldfrag_g(wih_lo, wB, 32);
      }
      #pragma unroll
      for (int j = 0; j < 4; ++j) {
        const int nt = g * 4 + j;
        acc[nt] = wmma_bf16(ahi, bh[j], acc[nt]);
        acc[nt] = wmma_bf16(alo, bh[j], acc[nt]);
        acc[nt] = wmma_bf16(ahi, bl[j], acc[nt]);
      }
    }
  }

  #pragma unroll
  for (int nt = 0; nt < PROJ_NT; ++nt) {
    const unsigned n = nbase + (unsigned)nt * 16u + lm;
    const float bias = bias_ih[n] + bias_hh[n];
    #pragma unroll
    for (int v = 0; v < 8; ++v) {
      const unsigned row  = mbase + (unsigned)v + 8u * lh;  // C layout: m = v + 8*lanehalf
      const unsigned oB   = (row << 12) + (n << 2);
      *(float*)((char*)out + (size_t)oB) = acc[nt][v] + bias;
    }
  }
}

// ---------------- grid barrier (sense-reversing, agent scope) ----------------
static __device__ __forceinline__ void grid_barrier(unsigned* cnt, unsigned* gen,
                                                    unsigned nwg) {
  __syncthreads();
  if (threadIdx.x == 0) {
    const unsigned g = __hip_atomic_load(gen, __ATOMIC_RELAXED, __HIP_MEMORY_SCOPE_AGENT);
    const unsigned arrived =
        __hip_atomic_fetch_add(cnt, 1u, __ATOMIC_ACQ_REL, __HIP_MEMORY_SCOPE_AGENT);
    if (arrived == nwg - 1u) {
      __hip_atomic_store(cnt, 0u, __ATOMIC_RELAXED, __HIP_MEMORY_SCOPE_AGENT);
      __hip_atomic_fetch_add(gen, 1u, __ATOMIC_ACQ_REL, __HIP_MEMORY_SCOPE_AGENT);
    } else {
      while (__hip_atomic_load(gen, __ATOMIC_ACQUIRE, __HIP_MEMORY_SCOPE_AGENT) == g) {
        __builtin_amdgcn_s_sleep(1);
      }
    }
  }
  __syncthreads();
}

// ---------------- phase 2: persistent cooperative scan ----------------
// 16 WGs x 256 threads. WG owns 64 hidden columns; wave = (ntile 0..3, khalf 0..1).
// W_hh hi/lo slices live in LDS (padded rows: 1024+8 elems). h hi/lo buffers are
// contiguous in ws behind ONE base pointer; ping-pong is a scalar 32-bit offset,
// so all h addressing is SADDR-form (no 64-bit VGPR address math per step).
#define RNN_LDSW  1032          // 1024 + 8 bf16 pad per row (bank-conflict avoidance)
#define RNN_LDS_BYTES (2 * 64 * RNN_LDSW * 2 + 4 * 256 * 4)   // whi + wlo + reduction
#define H_LO_OFF   32768u       // +32KB: lo plane within a ping-pong slot
#define H_SLOT_OFF 65536u       // +64KB: slot B

__global__ void __launch_bounds__(256)
rnn_scan_kernel(const float* __restrict__ w_hh, float* out,
                unsigned short* hbase,           // 4 contiguous 32KB planes
                unsigned* cnt, unsigned* gen) {
  extern __shared__ char smem[];
  bf16*  whi = (bf16*)smem;                       // [64][RNN_LDSW]
  bf16*  wlo = whi + 64 * RNN_LDSW;
  float* red = (float*)(smem + (size_t)2 * 64 * RNN_LDSW * 2);  // [4][256]

  const unsigned lane  = threadIdx.x & 31;
  const unsigned wave  = threadIdx.x >> 5;
  const unsigned lh    = lane >> 4;
  const unsigned lm    = lane & 15;
  const unsigned ntile = wave & 3;
  const unsigned khalf = wave >> 2;
  const unsigned nbase = blockIdx.x * 64;
  const unsigned n0    = nbase + ntile * 16;

  // --- preload + split W_hh slice into LDS (once) ---
  for (unsigned idx = threadIdx.x; idx < 64 * 256; idx += 256) {
    const unsigned row = idx >> 8;
    const unsigned c4  = (idx & 255) * 4;
    f32x4 v = *(const f32x4*)(w_hh + (size_t)(nbase + row) * 1024 + c4);
    bf16* dh = whi + row * RNN_LDSW + c4;
    bf16* dl = wlo + row * RNN_LDSW + c4;
    #pragma unroll
    for (int j = 0; j < 4; ++j) {
      bf16 h = (bf16)v[j];
      dh[j] = h;
      dl[j] = (bf16)(v[j] - (float)h);
    }
  }
  __syncthreads();

  const bf16* bb_hi = whi + (ntile * 16 + lm) * RNN_LDSW;  // B-frag row base (LDS)
  const bf16* bb_lo = wlo + (ntile * 16 + lm) * RNN_LDSW;

  const unsigned hRowB = lm << 11;           // lane's h row, *1024 bf16 *2B

  #pragma unroll 1
  for (unsigned t = 0; t < 1024; ++t) {
    const unsigned prevOff = (t & 1u) ? H_SLOT_OFF : 0u;       // read slot
    const unsigned nextOff = (t & 1u) ? 0u : H_SLOT_OFF;       // write slot
    f32x8 acc0 = zero8(), acc1 = zero8(), acc2 = zero8();
    const unsigned nIdx = n0 + lm;
    const unsigned tB   = t << 12;           // t*1024 floats *4B

    if (khalf == 0) {  // seed product-0 accumulator with x_proj(+biases) from d_out
      #pragma unroll
      for (int v = 0; v < 8; ++v) {
        const unsigned b  = (unsigned)v + 8u * lh;
        const unsigned oB = (b << 22) + tB + (nIdx << 2);   // ((b*1024+t)*1024+n)*4
        acc0[v] = *(const float*)((const char*)out + (size_t)oB);
      }
    }

    const int kc0 = khalf * 16;
    #pragma unroll
    for (int kc = kc0; kc < kc0 + 16; ++kc) {
      const unsigned kb  = (unsigned)kc * 32u;
      const unsigned aB  = prevOff + hRowB + ((kb + 8u * lh) << 1);
      bf16x16 ahi = ldfrag_g(hbase, aB, 32);
      bf16x16 alo = ldfrag_g(hbase, aB + H_LO_OFF, 32);
      bf16x16 bhi = ldfrag_lds(bb_hi + kb + 8 * lh, 16);
      bf16x16 blo = ldfrag_lds(bb_lo + kb + 8 * lh, 16);
      acc0 = wmma_bf16(ahi, bhi, acc0);   // three independent chains -> ILP
      acc1 = wmma_bf16(alo, bhi, acc1);
      acc2 = wmma_bf16(ahi, blo, acc2);
    }

    f32x8 s;
    #pragma unroll
    for (int v = 0; v < 8; ++v) s[v] = acc0[v] + acc1[v] + acc2[v];

    if (khalf == 1) {                    // publish partial sum (K upper half)
      float* r = red + ntile * 256 + lane * 8;
      #pragma unroll
      for (int v = 0; v < 8; ++v) r[v] = s[v];
    }
    __syncthreads();
    if (khalf == 0) {                    // combine, activate, store h_t
      const float* r = red + ntile * 256 + lane * 8;
      #pragma unroll
      for (int v = 0; v < 8; ++v) {
        const float    val = tanhf(s[v] + r[v]);
        const unsigned b   = (unsigned)v + 8u * lh;
        const unsigned oB  = (b << 22) + tB + (nIdx << 2);
        *(float*)((char*)out + (size_t)oB) = val;
        bf16 hh = (bf16)val;
        const unsigned hB = nextOff + (((b << 10) + nIdx) << 1);  // (b*1024+n)*2B
        *(bf16*)((char*)hbase + (size_t)hB) = hh;
        *(bf16*)((char*)hbase + (size_t)(hB + H_LO_OFF)) = (bf16)(val - (float)hh);
      }
    }

    grid_barrier(cnt, gen, gridDim.x);
  }
}

// ---------------- host launcher ----------------
extern "C" void kernel_launch(void* const* d_in, const int* in_sizes, int n_in,
                              void* d_out, int out_size, void* d_ws, size_t ws_size,
                              hipStream_t stream) {
  (void)in_sizes; (void)n_in; (void)out_size; (void)ws_size;
  const float* x       = (const float*)d_in[0];  // [16,1024,1024]
  const float* w_ih    = (const float*)d_in[1];  // [1024,1024]
  const float* w_hh    = (const float*)d_in[2];  // [1024,1024]
  const float* bias_ih = (const float*)d_in[3];  // [1024]
  const float* bias_hh = (const float*)d_in[4];  // [1024]
  float*       out     = (float*)d_out;          // [16,1024,1024]

  // workspace layout:
  //   [0]       barrier counter, [64] generation
  //   [256]     h planes: slotA_hi | slotA_lo | slotB_hi | slotB_lo  = 128 KB
  //   [131328]  W_ih split-bf16 hi plane (1024*1024 bf16)            = 2 MB
  //   [+2MB]    W_ih split-bf16 lo plane                             = 2 MB
  unsigned char* ws = (unsigned char*)d_ws;
  unsigned* cnt = (unsigned*)ws;
  unsigned* gen = (unsigned*)(ws + 64);
  unsigned short* hbase  = (unsigned short*)(ws + 256);
  unsigned short* wih_hi = (unsigned short*)(ws + 131328);
  unsigned short* wih_lo = wih_hi + 1024 * 1024;

  // zero barrier state + h0 (deterministic per call; graph-capturable memset node)
  hipMemsetAsync(d_ws, 0, 256 + 4 * 16 * 1024 * sizeof(unsigned short), stream);

  // Phase 0: one-shot split conversion of W_ih (1M elems, 4/thread)
  rnn_cvt_wih_kernel<<<dim3(1024), 256, 0, stream>>>(w_ih, wih_hi, wih_lo);

  // Phase 1: x_proj (+biases) into d_out
  rnn_proj_kernel<<<dim3(8, 128), 256, 0, stream>>>(x, wih_hi, wih_lo,
                                                    bias_ih, bias_hh, out);

  // Phase 2: persistent cooperative recurrent scan (16 WGs, one per WGP)
  rnn_scan_kernel<<<dim3(16), 256, RNN_LDS_BYTES, stream>>>(
      w_hh, out, hbase, cnt, gen);
}
